// Sequence_962072674603
// MI455X (gfx1250) — matrix operand order repaired and checked by reading, not compiled, not run on hardware
//
#include <hip/hip_runtime.h>

// ---------------------------------------------------------------------------
// 2-layer LSTM (D=50) + Linear(50->2), B=4096, T=1024, recurrent scan fused
// into one kernel. Grid: 128 WGs x 128 threads. Each WG owns 32 batch rows
// (2 row-groups of 16); waves split the gate/N dimension 2-ways, so each wave
// does 7 N-tiles x (2+4) K-steps = 42 WMMAs per step. Bias is folded into the
// GEMM via a constant-1.0 input column (accumulators start at inline C=0),
// and B-fragments are software-pipelined one tile ahead so LDS latency
// overlaps the WMMAs. ~140 KB LDS/WG -> 2 workgroups per 320 KB WGP.
// ---------------------------------------------------------------------------

typedef __attribute__((ext_vector_type(16))) __bf16        v16bf;
typedef __attribute__((ext_vector_type(8)))  float         v8f;
typedef __attribute__((ext_vector_type(4)))  unsigned int  v4u;

#define DD   50
#define GG   200                       // 4*D gate columns
#define TT   1024
#define BB   4096
#define NTP  14                        // padded N tiles (224 cols >= 200)
#define KS1  2                         // K=64  : [x(2); h1(50); 1.0; pad]
#define KS2  4                         // K=128 : [h1(50); h2(50); 1.0; pad]

// ---- LDS layout (bytes) ----------------------------------------------------
#define OFF_W1    0
#define W1_BYTES  (NTP*KS1*512*2)                  // 28672
#define OFF_W2    (OFF_W1 + W1_BYTES)
#define W2_BYTES  (NTP*KS2*512*2)                  // 57344
#define OFF_WLIN  (OFF_W2 + W2_BYTES)              // 86016
#define OFF_BLIN  (OFF_WLIN + 100*4)               // 86416
#define OFF_RG    ((OFF_BLIN + 8 + 15) & ~15)      // 86432
#define PW_A1     0                                // 16x64  bf16 = 2048 B
#define PW_A2     2048                             // 16x128 bf16 = 4096 B
#define PW_G      6144                             // 16x224 f32  = 14336 B
#define PW_C1     (PW_G + 16*224*4)                // 20480
#define PW_C2     (PW_C1 + 16*DD*4)                // 23680
#define PER_RG    (PW_C2 + 16*DD*4)                // 26880
#define SMEM_BYTES (OFF_RG + 2*PER_RG)             // 140192 B (2 WG / WGP)

__device__ __forceinline__ unsigned short f2bf(float f) {
    union { float f; unsigned int u; } x; x.f = f;
    unsigned int r = x.u + 0x7FFFu + ((x.u >> 16) & 1u);   // RNE
    return (unsigned short)(r >> 16);
}
__device__ __forceinline__ float bf2f(unsigned short h) {
    union { unsigned int u; float f; } x; x.u = ((unsigned int)h) << 16;
    return x.f;
}
// fast sigmoid/tanh on the v_exp/v_rcp transcendental pipes
__device__ __forceinline__ float sigm(float x) {
    return __builtin_amdgcn_rcpf(1.0f + __expf(-x));
}
__device__ __forceinline__ float tanh_(float x) {
    return 2.0f * __builtin_amdgcn_rcpf(1.0f + __expf(-2.0f * x)) - 1.0f;
}

union FragU { v16bf v; v4u q[2]; };
__device__ __forceinline__ v16bf load_frag(const unsigned short* p0,
                                           const unsigned short* p1) {
    FragU u;
    u.q[0] = *(const v4u*)p0;   // ds_load_b128
    u.q[1] = *(const v4u*)p1;   // ds_load_b128
    return u.v;
}
__device__ __forceinline__ v8f wmma_bf16(v16bf a, v16bf b, v8f c) {
    return __builtin_amdgcn_wmma_f32_16x16x32_bf16(
        false, a, false, b, (short)0, c, false, false);
}

__global__ __launch_bounds__(128)
void lstm2_wmma_kernel(const float* __restrict__ input,
                       const float* __restrict__ W_ih1, const float* __restrict__ W_hh1,
                       const float* __restrict__ b_ih1, const float* __restrict__ b_hh1,
                       const float* __restrict__ W_ih2, const float* __restrict__ W_hh2,
                       const float* __restrict__ b_ih2, const float* __restrict__ b_hh2,
                       const float* __restrict__ W_lin, const float* __restrict__ b_lin,
                       float* __restrict__ out) {
    extern __shared__ __align__(16) char smem[];
    const int tid  = threadIdx.x;
    const int wave = tid >> 5;
    const int lane = tid & 31;
    const int rg   = wave >> 1;        // row-group (16 rows) this wave computes
    const int ns   = wave & 1;         // N-split: tiles ns, ns+2, ... (7 tiles)

    unsigned short* w1   = (unsigned short*)(smem + OFF_W1);
    unsigned short* w2   = (unsigned short*)(smem + OFF_W2);
    float*          wlin = (float*)(smem + OFF_WLIN);
    float*          blin = (float*)(smem + OFF_BLIN);

    // ---- one-time: swizzle weights (+bias row) into B-fragment layout ------
    // block(nt,ks): 32 lanes x 16 bf16, lane-contiguous. lane l covers column
    // n = nt*16+(l&15); ushort j covers K = ks*32 + (l>>4)*16 + j.
    for (int idx = tid; idx < NTP * KS1 * 512; idx += 128) {
        int blk = idx >> 9, within = idx & 511;
        int l = within >> 4, j = within & 15;
        int nt = blk >> 1, ks = blk & 1;
        int n = nt * 16 + (l & 15);
        int k = ks * 32 + ((l >> 4) << 4) + j;
        float v = 0.0f;
        if (n < GG) {
            if (k < 2)        v = W_ih1[n * 2 + k];
            else if (k < 52)  v = W_hh1[n * DD + (k - 2)];
            else if (k == 52) v = b_ih1[n] + b_hh1[n];   // bias row (A col 52 = 1.0)
        }
        w1[idx] = f2bf(v);
    }
    for (int idx = tid; idx < NTP * KS2 * 512; idx += 128) {
        int blk = idx >> 9, within = idx & 511;
        int l = within >> 4, j = within & 15;
        int nt = blk >> 2, ks = blk & 3;
        int n = nt * 16 + (l & 15);
        int k = ks * 32 + ((l >> 4) << 4) + j;
        float v = 0.0f;
        if (n < GG) {
            if (k < DD)         v = W_ih2[n * DD + k];
            else if (k < 2*DD)  v = W_hh2[n * DD + (k - DD)];
            else if (k == 2*DD) v = b_ih2[n] + b_hh2[n]; // bias row (A col 100 = 1.0)
        }
        w2[idx] = f2bf(v);
    }
    for (int i = tid; i < 2 * DD; i += 128) wlin[i] = W_lin[i];
    if (tid < 2) blin[tid] = b_lin[tid];

    // ---- state init: h=c=0, constant-1.0 bias columns, zero padding --------
    for (int i = tid; i < 2 * 16 * 64; i += 128) {
        int g = i >> 10, o = i & 1023;
        ((unsigned short*)(smem + OFF_RG + g * PER_RG + PW_A1))[o] =
            ((o & 63) == 52) ? (unsigned short)0x3F80 : (unsigned short)0;
    }
    for (int i = tid; i < 2 * 16 * 128; i += 128) {
        int g = i >> 11, o = i & 2047;
        ((unsigned short*)(smem + OFF_RG + g * PER_RG + PW_A2))[o] =
            ((o & 127) == 100) ? (unsigned short)0x3F80 : (unsigned short)0;
    }
    for (int i = tid; i < 2 * 16 * DD; i += 128) {
        int g = (i >= 16 * DD), o = i - g * 16 * DD;
        ((float*)(smem + OFF_RG + g * PER_RG + PW_C1))[o] = 0.0f;
        ((float*)(smem + OFF_RG + g * PER_RG + PW_C2))[o] = 0.0f;
    }
    __syncthreads();

    // ---- per-role constants -------------------------------------------------
    unsigned short* a1w    = (unsigned short*)(smem + OFF_RG + rg * PER_RG + PW_A1);
    unsigned short* a2w    = (unsigned short*)(smem + OFF_RG + rg * PER_RG + PW_A2);
    float*          gatesw = (float*)(smem + OFF_RG + rg * PER_RG + PW_G);
    const int m     = lane & 15;          // A row
    const int kh    = (lane >> 4) << 4;   // A K-half (0/16)
    const int nlane = lane & 15;          // C/D column in tile
    const int mbase = (lane >> 4) << 3;   // C/D row base (0/8)
    const unsigned short* b1base = w1 + lane * 16;
    const unsigned short* b2base = w2 + lane * 16;

    // x / linear role (tid < 64): one (row, component) pair
    const int rowx = tid >> 1, comp = tid & 1;
    const int rgx  = rowx >> 4, rx = rowx & 15;
    unsigned short* a1x = (unsigned short*)(smem + OFF_RG + rgx * PER_RG + PW_A1);
    unsigned short* a2x = (unsigned short*)(smem + OFF_RG + rgx * PER_RG + PW_A2);
    const size_t xbase = ((size_t)(blockIdx.x * 32 + rowx) * TT) * 2 + comp;

    float xv = 0.0f;
    if (tid < 64) xv = input[xbase];      // prefetch x_0 into registers

    for (int t = 0; t < TT; ++t) {
        // --- phase 1: commit x_t to A1, start load of x_{t+1} ---------------
        if (tid < 64) {
            a1x[rx * 64 + comp] = f2bf(xv);
            if (t + 1 < TT) xv = input[xbase + (size_t)(t + 1) * 2];
        }
        __syncthreads();

        // --- phase 2: cell1 gates  [16,64] x [64,224]  (bias folded) --------
        {
            const unsigned short* ap = a1w + m * 64 + kh;
            v16bf A0 = load_frag(ap,      ap + 8);
            v16bf A1 = load_frag(ap + 32, ap + 40);
            // prime tile 0's B fragments, then pipeline one tile ahead
            const unsigned short* p0 = b1base + (ns * KS1 + 0) * 512;
            const unsigned short* p1 = b1base + (ns * KS1 + 1) * 512;
            v16bf B0 = load_frag(p0, p0 + 8);
            v16bf B1 = load_frag(p1, p1 + 8);
#pragma unroll
            for (int i = 0; i < 7; ++i) {
                int nt = ns + 2 * i;
                v16bf nB0, nB1;
                if (i < 6) {
                    const unsigned short* q0 = b1base + ((nt + 2) * KS1 + 0) * 512;
                    const unsigned short* q1 = b1base + ((nt + 2) * KS1 + 1) * 512;
                    nB0 = load_frag(q0, q0 + 8);
                    nB1 = load_frag(q1, q1 + 8);
                }
                v8f acc = {};
                acc = wmma_bf16(A0, B0, acc);
                acc = wmma_bf16(A1, B1, acc);
                int n = nt * 16 + nlane;
#pragma unroll
                for (int j = 0; j < 8; ++j) gatesw[(mbase + j) * 224 + n] = acc[j];
                if (i < 6) { B0 = nB0; B1 = nB1; }
            }
        }
        __syncthreads();

        // --- phase 3: cell1 elementwise over 32x50 ---------------------------
        for (int idx = tid; idx < 32 * DD; idx += 128) {
            int row = idx / DD, u = idx - row * DD;
            int g = row >> 4, r = row & 15;
            char* rb = smem + OFF_RG + g * PER_RG;
            const float* gr = (const float*)(rb + PW_G) + r * 224;
            float* cc = (float*)(rb + PW_C1) + r * DD + u;
            float gi = gr[u], gf = gr[DD + u], gg = gr[2*DD + u], go = gr[3*DD + u];
            float cn = sigm(gf) * cc[0] + sigm(gi) * tanh_(gg);
            float hn = sigm(go) * tanh_(cn);
            cc[0] = cn;
            unsigned short hb = f2bf(hn);
            ((unsigned short*)(rb + PW_A1))[r * 64 + 2 + u] = hb;
            ((unsigned short*)(rb + PW_A2))[r * 128 + u]    = hb;
        }
        __syncthreads();

        // --- phase 4: cell2 gates  [16,128] x [128,224] ----------------------
        {
            const unsigned short* ap = a2w + m * 128 + kh;
            v16bf A[KS2];
#pragma unroll
            for (int ks = 0; ks < KS2; ++ks)
                A[ks] = load_frag(ap + ks * 32, ap + ks * 32 + 8);
            v16bf B[KS2];
#pragma unroll
            for (int ks = 0; ks < KS2; ++ks) {
                const unsigned short* p = b2base + (ns * KS2 + ks) * 512;
                B[ks] = load_frag(p, p + 8);
            }
#pragma unroll
            for (int i = 0; i < 7; ++i) {
                int nt = ns + 2 * i;
                v16bf nB[KS2];
                if (i < 6) {
#pragma unroll
                    for (int ks = 0; ks < KS2; ++ks) {
                        const unsigned short* q = b2base + ((nt + 2) * KS2 + ks) * 512;
                        nB[ks] = load_frag(q, q + 8);
                    }
                }
                v8f acc = {};
#pragma unroll
                for (int ks = 0; ks < KS2; ++ks)
                    acc = wmma_bf16(A[ks], B[ks], acc);
                int n = nt * 16 + nlane;
#pragma unroll
                for (int j = 0; j < 8; ++j) gatesw[(mbase + j) * 224 + n] = acc[j];
                if (i < 6) {
#pragma unroll
                    for (int ks = 0; ks < KS2; ++ks) B[ks] = nB[ks];
                }
            }
        }
        __syncthreads();

        // --- phase 5: cell2 elementwise --------------------------------------
        for (int idx = tid; idx < 32 * DD; idx += 128) {
            int row = idx / DD, u = idx - row * DD;
            int g = row >> 4, r = row & 15;
            char* rb = smem + OFF_RG + g * PER_RG;
            const float* gr = (const float*)(rb + PW_G) + r * 224;
            float* cc = (float*)(rb + PW_C2) + r * DD + u;
            float gi = gr[u], gf = gr[DD + u], gg = gr[2*DD + u], go = gr[3*DD + u];
            float cn = sigm(gf) * cc[0] + sigm(gi) * tanh_(gg);
            float hn = sigm(go) * tanh_(cn);
            cc[0] = cn;
            ((unsigned short*)(rb + PW_A2))[r * 128 + DD + u] = f2bf(hn);
        }
        __syncthreads();

        // --- phase 6: linear 50->2 from bf16 h2 + store (no trailing barrier;
        //     next write to these a2 cells is 4 barriers away) ----------------
        if (tid < 64) {
            float s = blin[comp];
            const unsigned short* hp = a2x + rx * 128 + DD;
            const float* wp = wlin + comp * DD;
#pragma unroll 10
            for (int u = 0; u < DD; ++u) s += bf2f(hp[u]) * wp[u];
            out[xbase + (size_t)t * 2] = s;
        }
    }
}

extern "C" void kernel_launch(void* const* d_in, const int* in_sizes, int n_in,
                              void* d_out, int out_size, void* d_ws, size_t ws_size,
                              hipStream_t stream) {
    (void)in_sizes; (void)n_in; (void)out_size; (void)d_ws; (void)ws_size;
    const float* input = (const float*)d_in[0];
    const float* W_ih1 = (const float*)d_in[1];
    const float* W_hh1 = (const float*)d_in[2];
    const float* b_ih1 = (const float*)d_in[3];
    const float* b_hh1 = (const float*)d_in[4];
    const float* W_ih2 = (const float*)d_in[5];
    const float* W_hh2 = (const float*)d_in[6];
    const float* b_ih2 = (const float*)d_in[7];
    const float* b_hh2 = (const float*)d_in[8];
    const float* W_lin = (const float*)d_in[9];
    const float* b_lin = (const float*)d_in[10];
    float* out = (float*)d_out;

    hipFuncSetAttribute((const void*)lstm2_wmma_kernel,
                        hipFuncAttributeMaxDynamicSharedMemorySize, SMEM_BYTES);

    dim3 grid(BB / 32);    // 128 workgroups, each owns 32 batch rows
    dim3 block(128);       // 4 waves: 2 row-groups x 2 N-splits
    lstm2_wmma_kernel<<<grid, block, SMEM_BYTES, stream>>>(
        input, W_ih1, W_hh1, b_ih1, b_hh1,
        W_ih2, W_hh2, b_ih2, b_hh2, W_lin, b_lin, out);
}
